// VarianceAdaptor_2499670966506
// MI455X (gfx1250) — compile-verified, matrix operand order checked by reading, not run on hardware
//
#include <hip/hip_runtime.h>
#include <hip/hip_bf16.h>

// ---------------------------------------------------------------------------
// Problem constants (from the reference)
// ---------------------------------------------------------------------------
#define BB   32
#define NN   512
#define TT   2048
#define DD   256      // input channels (== F)
#define FF   256      // conv channels
#define KK   3        // conv taps
#define DK   (DD*KK)  // 768, reference w layout is (F, D, K): w[f*DK + d*KK + k]
#define WELEM (KK*FF*DD)   // 196608 elements per repacked conv weight

typedef __attribute__((ext_vector_type(16))) __bf16           v16bf;
typedef __attribute__((ext_vector_type(8)))  float            v8f;
typedef __attribute__((ext_vector_type(4)))  unsigned short   u16x4;
typedef __attribute__((ext_vector_type(8)))  unsigned short   u16x8;
typedef __attribute__((ext_vector_type(16))) unsigned short   u16x16;

// float -> bf16 (round to nearest even)
__device__ __forceinline__ unsigned short f2bf(float f) {
    unsigned int x = __builtin_bit_cast(unsigned int, f);
    x += 0x7FFFu + ((x >> 16) & 1u);
    return (unsigned short)(x >> 16);
}

// Load a 16x32 (A) or 32x16 (B) bf16 WMMA fragment from an LDS row.
// CDNA5 wave32 layout: element e maps to K = d0 + e + 8*((e>>3) + hi)
// => two contiguous 8x-ushort (16B) chunks at d0+8*hi and d0+16+8*hi.
__device__ __forceinline__ v16bf load_frag(const unsigned short* rowp, int d0, int hi) {
    const u16x8 c0 = *(const u16x8*)(rowp + d0 + 8 * hi);
    const u16x8 c1 = *(const u16x8*)(rowp + d0 + 16 + 8 * hi);
    u16x16 t;
#pragma unroll
    for (int i = 0; i < 8; ++i) { t[i] = c0[i]; t[i + 8] = c1[i]; }
    return __builtin_bit_cast(v16bf, t);
}

// ---------------------------------------------------------------------------
// One-time weight repack: w(F,D,K) f32  ->  wbf[k][f][d] bf16
// ---------------------------------------------------------------------------
__global__ void __launch_bounds__(256)
repack_w(const float* __restrict__ w, unsigned short* __restrict__ wbf) {
    const int idx = blockIdx.x * 256 + threadIdx.x;   // < KK*FF*DD
    const int k = idx >> 16;          // FF*DD == 65536 per tap
    const int f = (idx >> 8) & 255;
    const int d = idx & 255;
    wbf[idx] = f2bf(w[(size_t)f * DK + d * KK + k]);
}

// ---------------------------------------------------------------------------
// Conv1d(K=3, pad=1) as WMMA GEMM, fused bias + ReLU.
//   out[r, f] = relu( sum_{k,d} in[r + k - 1, d] * w[f, d, k] + bias[f] )
// Block tile: 64 rows x ALL 256 channels.  8 waves = 4(M subtile) x 2(F half);
// each wave owns 16 rows x 128 channels = 8 16x16 f32 accumulators, so one
// A fragment feeds 8 v_wmma_f32_16x16x32_bf16 ops.
// grid.x = rows/64. Rows never cross a batch boundary (L % 64 == 0);
// conv halo rows zero-padded per sequence.
// ---------------------------------------------------------------------------
#define APITCH 264   // 256 + 8 ushort pad: row = 528B (16B aligned, bank-staggered)
#define BPITCH 40    // 32 + 8 ushort pad: row = 80B  (16B aligned, conflict-free)

__global__ void __launch_bounds__(256)
conv_wmma_relu(const float* __restrict__ in, const unsigned short* __restrict__ wbf,
               const float* __restrict__ bias, float* __restrict__ outp, int L) {
    __shared__ __align__(16) unsigned short sA[66 * APITCH];  // 34,848 B
    __shared__ __align__(16) unsigned short sB[256 * BPITCH]; // 20,480 B

    const int rowBase = blockIdx.x * 64;
    const int n0      = rowBase % L;            // position within the sequence
    const int seqBase = rowBase - n0;           // first row of this batch

    // ---- stage A tile (f32 -> bf16, float4-vectorized), zero-pad outside [0,L) ----
    for (int idx = threadIdx.x; idx < 66 * 64; idx += 256) {
        const int r  = idx >> 6;
        const int c4 = (idx & 63) << 2;
        const int n  = n0 + r - 1;
        float4 v = make_float4(0.f, 0.f, 0.f, 0.f);
        if (n >= 0 && n < L) v = *(const float4*)(in + (size_t)(seqBase + n) * DD + c4);
        u16x4 o;
        o[0] = f2bf(v.x); o[1] = f2bf(v.y); o[2] = f2bf(v.z); o[3] = f2bf(v.w);
        *(u16x4*)(sA + r * APITCH + c4) = o;
    }

    const int wave = threadIdx.x >> 5;
    const int lane = threadIdx.x & 31;
    const int lm   = lane & 15;
    const int hi   = lane >> 4;
    const int mi   = wave & 3;            // M subtile (16 rows each)
    const int fh   = (wave >> 2) * 128;   // F half owned by this wave

    v8f acc[8];
#pragma unroll
    for (int t = 0; t < 8; ++t) acc[t] = (v8f){};

    for (int k = 0; k < KK; ++k) {
        for (int dc = 0; dc < 8; ++dc) {
            const int d0 = dc * 32;
            __syncthreads();
            // ---- stage bf16 weight chunk: sB[f][0..31] = wbf[k][f][d0..d0+31]
            // one 64B contiguous row-slice per thread, pure vector copy ----
            {
                const int f = threadIdx.x;
                const unsigned short* srow = wbf + ((size_t)k * FF + f) * DD + d0;
                u16x8* drow = (u16x8*)(sB + f * BPITCH);
#pragma unroll
                for (int j = 0; j < 4; ++j) drow[j] = *(const u16x8*)(srow + j * 8);
            }
            __syncthreads();

            const v16bf a = load_frag(sA + (mi * 16 + lm + k) * APITCH, d0, hi);
#pragma unroll
            for (int t = 0; t < 8; ++t) {
                const v16bf b = load_frag(sB + (fh + t * 16 + lm) * BPITCH, 0, hi);
                acc[t] = __builtin_amdgcn_wmma_f32_16x16x32_bf16(
                    false, a, false, b, (short)0, acc[t], false, false);
            }
        }
    }

    // ---- epilogue: bias + ReLU, scatter D fragments (M = r + 8*hi, N = lm) ----
    const int rowSub = rowBase + mi * 16 + 8 * hi;
#pragma unroll
    for (int t = 0; t < 8; ++t) {
        const int f = fh + t * 16 + lm;
        const float bv = bias[f];
#pragma unroll
        for (int r = 0; r < 8; ++r) {
            float v = acc[t][r] + bv;
            v = v > 0.f ? v : 0.f;
            outp[(size_t)(rowSub + r) * FF + f] = v;
        }
    }
}

// ---------------------------------------------------------------------------
// LayerNorm over F=256, in place. One wave32 per row, 8 rows per block.
// ---------------------------------------------------------------------------
__global__ void __launch_bounds__(256)
ln_kernel(float* __restrict__ h, const float* __restrict__ g,
          const float* __restrict__ be, int M) {
    const int wave = threadIdx.x >> 5;
    const int lane = threadIdx.x & 31;
    const int row  = blockIdx.x * 8 + wave;
    if (row >= M) return;
    float* p = h + (size_t)row * FF;

    float v[8];
    float s = 0.f, s2 = 0.f;
#pragma unroll
    for (int i = 0; i < 8; ++i) {
        v[i] = p[lane + 32 * i];
        s  += v[i];
        s2 += v[i] * v[i];
    }
#pragma unroll
    for (int off = 16; off >= 1; off >>= 1) {
        s  += __shfl_xor(s,  off, 32);
        s2 += __shfl_xor(s2, off, 32);
    }
    const float mu   = s * (1.f / FF);
    const float var  = s2 * (1.f / FF) - mu * mu;
    const float rstd = rsqrtf(var + 1e-5f);
#pragma unroll
    for (int i = 0; i < 8; ++i) {
        const int fidx = lane + 32 * i;
        p[fidx] = (v[i] - mu) * rstd * g[fidx] + be[fidx];
    }
}

// ---------------------------------------------------------------------------
// Final projection F -> 1. One wave per row. Writes out1 (and out2 if non-null).
// ---------------------------------------------------------------------------
__global__ void __launch_bounds__(256)
fc_kernel(const float* __restrict__ h, const float* __restrict__ fcw,
          const float* __restrict__ fcb, float* __restrict__ out1,
          float* __restrict__ out2, int M) {
    const int wave = threadIdx.x >> 5;
    const int lane = threadIdx.x & 31;
    const int row  = blockIdx.x * 8 + wave;
    if (row >= M) return;
    const float* p = h + (size_t)row * FF;

    float s = 0.f;
#pragma unroll
    for (int i = 0; i < 8; ++i) {
        const int fidx = lane + 32 * i;
        s += p[fidx] * fcw[fidx];
    }
#pragma unroll
    for (int off = 16; off >= 1; off >>= 1) s += __shfl_xor(s, off, 32);
    if (lane == 0) {
        const float val = s + fcb[0];
        out1[row] = val;
        if (out2) out2[row] = val;
    }
}

// ---------------------------------------------------------------------------
// Per-batch inclusive cumsum of durations (B=32 serial scans of N=512).
// ---------------------------------------------------------------------------
__global__ void cumsum_kernel(const int* __restrict__ dur, int* __restrict__ cum) {
    const int b = threadIdx.x;
    if (b >= BB) return;
    int s = 0;
    for (int n = 0; n < NN; ++n) {
        s += dur[b * NN + n];
        cum[b * NN + n] = s;
    }
}

// ---------------------------------------------------------------------------
// Length regulation: x_exp[b,t,:] = valid(t) ? x[b, min(searchsorted_right(cum,t), N-1), :] : 0
// 64 lanes per (b,t) row, 4 rows per block. float4 copies.
// ---------------------------------------------------------------------------
__global__ void __launch_bounds__(256)
gather_kernel(const float* __restrict__ x, const int* __restrict__ cum,
              float* __restrict__ xe) {
    const int sub  = threadIdx.x >> 6;
    const int lane = threadIdx.x & 63;
    const int row  = blockIdx.x * 4 + sub;   // in [0, B*T)
    const int b    = row >> 11;              // T = 2048
    const int t    = row & (TT - 1);

    const int* c = cum + b * NN;
    int lo = 0, hiB = NN;
    while (lo < hiB) {
        const int mid = (lo + hiB) >> 1;
        if (c[mid] <= t) lo = mid + 1; else hiB = mid;
    }
    const bool valid = t < c[NN - 1];
    const int idx = lo < (NN - 1) ? lo : (NN - 1);

    float4 v = make_float4(0.f, 0.f, 0.f, 0.f);
    if (valid) v = *(const float4*)(x + ((size_t)b * NN + idx) * DD + lane * 4);
    *(float4*)(xe + (size_t)row * DD + lane * 4) = v;
}

// ---------------------------------------------------------------------------
// out[b,t,:] += pitch[b,t] + energy[b,t]   (out already holds masked x_exp)
// ---------------------------------------------------------------------------
__global__ void __launch_bounds__(256)
final_add_kernel(float* __restrict__ outp, const float* __restrict__ pitch,
                 const float* __restrict__ energy) {
    const int sub  = threadIdx.x >> 6;
    const int lane = threadIdx.x & 63;
    const int row  = blockIdx.x * 4 + sub;
    const float s  = pitch[row] + energy[row];
    float4* p = (float4*)(outp + (size_t)row * DD + lane * 4);
    float4 v = *p;
    v.x += s; v.y += s; v.z += s; v.w += s;
    *p = v;
}

// ---------------------------------------------------------------------------
// Full predictor: conv+relu -> LN -> conv+relu -> LN -> fc
// ---------------------------------------------------------------------------
static void run_predictor(const float* inp, const unsigned short* wbf1,
                          const unsigned short* wbf2, const float* const* P,
                          float* h0, float* h1, float* out1, float* out2,
                          int M, int L, hipStream_t s) {
    conv_wmma_relu<<<M / 64, 256, 0, s>>>(inp, wbf1, P[1], h0, L);
    ln_kernel<<<M / 8, 256, 0, s>>>(h0, P[2], P[3], M);
    conv_wmma_relu<<<M / 64, 256, 0, s>>>(h0, wbf2, P[5], h1, L);
    ln_kernel<<<M / 8, 256, 0, s>>>(h1, P[6], P[7], M);
    fc_kernel<<<M / 8, 256, 0, s>>>(h1, P[8], P[9], out1, out2, M);
}

extern "C" void kernel_launch(void* const* d_in, const int* in_sizes, int n_in,
                              void* d_out, int out_size, void* d_ws, size_t ws_size,
                              hipStream_t stream) {
    (void)in_sizes; (void)n_in; (void)out_size; (void)ws_size;

    // ---- inputs (setup_inputs dict order; param dicts flatten leaf-by-leaf) ----
    const float* x   = (const float*)d_in[0];
    const int*   dur = (const int*)d_in[1];
    // d_in[2] = mel_max_length (compile-time TT)
    const float* dp[10], *pp[10], *ep[10];
    for (int i = 0; i < 10; ++i) {
        dp[i] = (const float*)d_in[3 + i];
        pp[i] = (const float*)d_in[13 + i];
        ep[i] = (const float*)d_in[23 + i];
    }

    // ---- output layout: out(B*T*D) | pos(B*N) | duration(B*N) | pitch(B*T) | energy(B*T)
    float* out      = (float*)d_out;
    float* pos      = out + (size_t)BB * TT * DD;
    float* duration = pos + (size_t)BB * NN;
    float* pitch    = duration + (size_t)BB * NN;
    float* energy   = pitch + (size_t)BB * TT;

    // ---- workspace: two activation buffers + cumsum + 6 repacked bf16 weights ----
    float* h0 = (float*)d_ws;
    float* h1 = h0 + (size_t)BB * TT * FF;
    int*  cum = (int*)(h1 + (size_t)BB * TT * FF);
    unsigned short* wb = (unsigned short*)(cum + BB * NN);
    unsigned short* w_d1 = wb;                 // dur conv1
    unsigned short* w_d2 = wb + 1 * WELEM;     // dur conv2
    unsigned short* w_p1 = wb + 2 * WELEM;
    unsigned short* w_p2 = wb + 3 * WELEM;
    unsigned short* w_e1 = wb + 4 * WELEM;
    unsigned short* w_e2 = wb + 5 * WELEM;

    const int Mdur = BB * NN;   // 16384
    const int Mexp = BB * TT;   // 65536

    // 0) one-time weight repack to bf16 [k][f][d] (L2-resident thereafter)
    repack_w<<<WELEM / 256, 256, 0, stream>>>(dp[0], w_d1);
    repack_w<<<WELEM / 256, 256, 0, stream>>>(dp[4], w_d2);
    repack_w<<<WELEM / 256, 256, 0, stream>>>(pp[0], w_p1);
    repack_w<<<WELEM / 256, 256, 0, stream>>>(pp[4], w_p2);
    repack_w<<<WELEM / 256, 256, 0, stream>>>(ep[0], w_e1);
    repack_w<<<WELEM / 256, 256, 0, stream>>>(ep[4], w_e2);

    // 1) duration predictor on x; writes both pos and duration slots
    run_predictor(x, w_d1, w_d2, dp, h0, h1, duration, pos, Mdur, NN, stream);

    // 2) length regulation -> x_exp written directly into the `out` slot
    cumsum_kernel<<<1, 32, 0, stream>>>(dur, cum);
    gather_kernel<<<Mexp / 4, 256, 0, stream>>>(x, cum, out);

    // 3) pitch / energy predictors on x_exp (read from `out`, still pure x_exp)
    run_predictor(out, w_p1, w_p2, pp, h0, h1, pitch,  nullptr, Mexp, TT, stream);
    run_predictor(out, w_e1, w_e2, ep, h0, h1, energy, nullptr, Mexp, TT, stream);

    // 4) out = x_exp + pitch + energy
    final_add_kernel<<<Mexp / 4, 256, 0, stream>>>(out, pitch, energy);
}
